// EPA_34832184771359
// MI455X (gfx1250) — matrix operand order repaired and checked by reading, not compiled, not run on hardware
//
#include <hip/hip_runtime.h>

// ---------------------------------------------------------------------------
// EPA dual-attention forward for MI455X (gfx1250, wave32, WMMA).
// All contractions run as bf16 WMMA GEMMs (v_wmma_f32_16x16x32_bf16) with f32
// accumulation; softmax / l2-normalize stay in f32.
// GEMM: 128x64 block tile, 8 waves, 32x32 per-wave tile (4 WMMA accumulators),
// gfx1250 async global->LDS staging (ASYNCcnt) via inline asm.
// ---------------------------------------------------------------------------

#define BB 8
#define NN 4096
#define CC 256
#define HH 4
#define DD 64
#define PP 64
#define MT (BB * NN)          // 32768 token rows

// Toggle the CDNA5 async-DMA staging path (inline asm). Fallback = VGPR staging.
#define USE_ASYNC_LDS 1

typedef __bf16 bf16;
typedef __attribute__((ext_vector_type(16))) __bf16          bf16x16;
typedef __attribute__((ext_vector_type(8)))  float           f32x8;
typedef __attribute__((ext_vector_type(8)))  unsigned short  u16x8;
typedef __attribute__((ext_vector_type(16))) unsigned short  u16x16;

__device__ __forceinline__ unsigned short f32_bf16(float f) {
    union { float f; unsigned u; } v; v.f = f;
    unsigned u = v.u;
    u += 0x7FFFu + ((u >> 16) & 1u);            // round-to-nearest-even
    return (unsigned short)(u >> 16);
}

#ifdef USE_ASYNC_LDS
// gfx1250: async DMA global -> LDS, 16B per lane, tracked by ASYNCcnt.
// vdst VGPR carries the LDS byte address (generic LDS ptr truncates to it).
__device__ __forceinline__ void async_copy_b128(const unsigned short* g, unsigned short* l) {
    unsigned lds = (unsigned)(unsigned long long)(void*)l;
    asm volatile("global_load_async_to_lds_b128 %0, %1, off"
                 :: "v"(lds), "v"(g)
                 : "memory");
}
__device__ __forceinline__ void wait_async0() {
    asm volatile("s_wait_asynccnt 0x0" ::: "memory");
}
#endif

// ---------------------------------------------------------------------------
// Batched bf16 GEMM:  C[bi][m][n] = sum_k A[bi][m][k] * B[bi][n][k]  (+bias,+PReLU)
// A,B bf16 (ushort), row-major, contiguous K. Requires: Ncols % 64 == 0,
// K % 64 == 0, lda/ldb % 8 == 0. M edge-guarded (any multiple of 16).
// C batch offset = (bi/cInnerMod)*cStrideOuter + (bi%cInnerMod)*cStrideInner.
// ---------------------------------------------------------------------------
__global__ __launch_bounds__(256)
void gemm_bf16_wmma(const unsigned short* __restrict__ A, long long aStride, int lda,
                    const unsigned short* __restrict__ Bm, long long bStride, int ldb,
                    float* __restrict__ C, long long cStrideOuter, long long cStrideInner,
                    int cInnerMod, int ldc, int cColOff,
                    const float* __restrict__ bias,
                    const float* __restrict__ prelu,
                    int M, int Ncols, int K)
{
    __shared__ __align__(32) unsigned short As[128 * 64];   // 16 KB
    __shared__ __align__(32) unsigned short Bs[64 * 64];    //  8 KB

    const int t    = threadIdx.x;
    const int w    = t >> 5;
    const int lane = t & 31;
    const int bi   = blockIdx.z;

    const int rowBlk = blockIdx.x * 128;
    const int colBlk = blockIdx.y * 64;

    const unsigned short* Ab = A  + (long long)bi * aStride;
    const unsigned short* Bb = Bm + (long long)bi * bStride;

    // staging maps: A tile 128x64 -> 32 elems/thread; B tile 64x64 -> 16/thread
    const int lrA = t >> 1;            // 0..127
    const int lcA = (t & 1) * 32;      // 0,32
    const int lrB = t >> 2;            // 0..63
    const int lcB = (t & 3) * 16;      // 0,16,32,48
    const int grA = (rowBlk + lrA < M) ? (rowBlk + lrA) : (M - 1);  // clamp edge

    // wave tiling: 4 row-strips (32 rows) x 2 col-strips (32 cols)
    const int tileRow = (w & 3) * 32;
    const int wcol    = (w >> 2) * 32;

    // Lane layout used here: contiguous k = 16*(lane/16) + s (16-bit A 16x32 / B 32x16, wave32)
    const int mfrag = lane & 15;
    const int ka    = (lane < 16) ? 0 : 16;   // A: K half start
    const int kb16  = (lane < 16) ? 0 : 16;   // B: K half start

    f32x8 acc00 = {}, acc01 = {}, acc10 = {}, acc11 = {};

    for (int kbase = 0; kbase < K; kbase += 64) {
        const unsigned short* ag = Ab + (long long)grA * lda + kbase + lcA;
        const unsigned short* bg = Bb + (long long)(colBlk + lrB) * ldb + kbase + lcB;
#ifdef USE_ASYNC_LDS
        async_copy_b128(ag,      &As[lrA * 64 + lcA]);
        async_copy_b128(ag + 8,  &As[lrA * 64 + lcA + 8]);
        async_copy_b128(ag + 16, &As[lrA * 64 + lcA + 16]);
        async_copy_b128(ag + 24, &As[lrA * 64 + lcA + 24]);
        async_copy_b128(bg,      &Bs[lrB * 64 + lcB]);
        async_copy_b128(bg + 8,  &Bs[lrB * 64 + lcB + 8]);
        wait_async0();
#else
        *(u16x8*)&As[lrA * 64 + lcA]      = *(const u16x8*)ag;
        *(u16x8*)&As[lrA * 64 + lcA + 8]  = *(const u16x8*)(ag + 8);
        *(u16x8*)&As[lrA * 64 + lcA + 16] = *(const u16x8*)(ag + 16);
        *(u16x8*)&As[lrA * 64 + lcA + 24] = *(const u16x8*)(ag + 24);
        *(u16x8*)&Bs[lrB * 64 + lcB]      = *(const u16x8*)bg;
        *(u16x8*)&Bs[lrB * 64 + lcB + 8]  = *(const u16x8*)(bg + 8);
#endif
        if (kbase + 64 < K) __builtin_prefetch(ag + 64, 0, 1);   // global_prefetch_b8
        __syncthreads();

        #pragma unroll
        for (int kk = 0; kk < 64; kk += 32) {
            // A fragments: lane<16 -> M=lane, K=0..15; lane>=16 -> K=16..31
            const int a0 = (tileRow + mfrag) * 64 + kk + ka;
            u16x8 a0lo = *(const u16x8*)&As[a0];
            u16x8 a0hi = *(const u16x8*)&As[a0 + 8];
            bf16x16 av0 = __builtin_bit_cast(bf16x16,
                (u16x16)__builtin_shufflevector(a0lo, a0hi,
                    0,1,2,3,4,5,6,7,8,9,10,11,12,13,14,15));
            const int a1 = (tileRow + 16 + mfrag) * 64 + kk + ka;
            u16x8 a1lo = *(const u16x8*)&As[a1];
            u16x8 a1hi = *(const u16x8*)&As[a1 + 8];
            bf16x16 av1 = __builtin_bit_cast(bf16x16,
                (u16x16)__builtin_shufflevector(a1lo, a1hi,
                    0,1,2,3,4,5,6,7,8,9,10,11,12,13,14,15));

            // B fragments: lane<16 -> N=lane, K=0..15; lane>=16 -> K=16..31
            u16x16 b0u = *(const u16x16*)&Bs[(wcol + mfrag) * 64 + kk + kb16];
            u16x16 b1u = *(const u16x16*)&Bs[(wcol + 16 + mfrag) * 64 + kk + kb16];
            bf16x16 bv0 = __builtin_bit_cast(bf16x16, b0u);
            bf16x16 bv1 = __builtin_bit_cast(bf16x16, b1u);

            acc00 = __builtin_amdgcn_wmma_f32_16x16x32_bf16(
                        false, av0, false, bv0, (short)0, acc00, false, false);
            acc01 = __builtin_amdgcn_wmma_f32_16x16x32_bf16(
                        false, av0, false, bv1, (short)0, acc01, false, false);
            acc10 = __builtin_amdgcn_wmma_f32_16x16x32_bf16(
                        false, av1, false, bv0, (short)0, acc10, false, false);
            acc11 = __builtin_amdgcn_wmma_f32_16x16x32_bf16(
                        false, av1, false, bv1, (short)0, acc11, false, false);
        }
        __syncthreads();
    }

    // epilogue per ISA C layout: VGPR i -> M = base + i (+8 for lanes 16..31)
    const long long cBase = (long long)(bi / cInnerMod) * cStrideOuter
                          + (long long)(bi % cInnerMod) * cStrideInner;
    const int mBase = rowBlk + tileRow + ((lane < 16) ? 0 : 8);
    const int n0 = colBlk + wcol + (lane & 15);
    const int n1 = n0 + 16;
    const float slope = prelu ? *prelu : 0.f;
    const float bias0 = bias ? bias[n0] : 0.f;
    const float bias1 = bias ? bias[n1] : 0.f;

    #pragma unroll
    for (int i = 0; i < 8; ++i) {
        const int r0 = mBase + i;         // row fragment 0
        const int r1 = r0 + 16;           // row fragment 1
        float v00 = acc00[i] + bias0, v01 = acc01[i] + bias1;
        float v10 = acc10[i] + bias0, v11 = acc11[i] + bias1;
        if (prelu) {
            v00 = (v00 >= 0.f) ? v00 : slope * v00;
            v01 = (v01 >= 0.f) ? v01 : slope * v01;
            v10 = (v10 >= 0.f) ? v10 : slope * v10;
            v11 = (v11 >= 0.f) ? v11 : slope * v11;
        }
        if (r0 < M) {
            C[cBase + (long long)r0 * ldc + n0 + cColOff] = v00;
            C[cBase + (long long)r0 * ldc + n1 + cColOff] = v01;
        }
        if (r1 < M) {
            C[cBase + (long long)r1 * ldc + n0 + cColOff] = v10;
            C[cBase + (long long)r1 * ldc + n1 + cColOff] = v11;
        }
    }
}

// ---------------------------------------------------------------------------
// elementwise f32 -> bf16
// ---------------------------------------------------------------------------
__global__ __launch_bounds__(256)
void cvt_f32_bf16(const float* __restrict__ in, unsigned short* __restrict__ out, long long n)
{
    long long i = (long long)blockIdx.x * 256 + threadIdx.x;
    if (i < n) out[i] = f32_bf16(in[i]);
}

// ---------------------------------------------------------------------------
// inverse L2 norms over tokens for q,k rows: inv[((comp*B+b)*H+h)*D+dd]
// ---------------------------------------------------------------------------
__global__ __launch_bounds__(256)
void qk_invnorm(const float* __restrict__ qkvv, float* __restrict__ inv)
{
    int id = blockIdx.x;
    const int dd = id % DD;  id /= DD;
    const int h  = id % HH;  id /= HH;
    const int b  = id % BB;
    const int comp = id / BB;

    float s = 0.f;
    for (int n = threadIdx.x; n < NN; n += 256) {
        float v = qkvv[((((long long)b * NN + n) * 4 + comp) * HH + h) * DD + dd];
        s += v * v;
    }
    __shared__ float red[256];
    red[threadIdx.x] = s;
    __syncthreads();
    for (int o = 128; o > 0; o >>= 1) {
        if (threadIdx.x < o) red[threadIdx.x] += red[threadIdx.x + o];
        __syncthreads();
    }
    if (threadIdx.x == 0)
        inv[blockIdx.x] = 1.f / fmaxf(sqrtf(red[0]), 1e-12f);
}

// ---------------------------------------------------------------------------
// fan qkvv out into the bf16 layouts the GEMMs want:
//   qn_t,kn_t,k_t,vsa_t : [BH][D][N]  (K = tokens, contiguous)
//   qn_tok,vca_tok      : [BH][N][D]  (K = channels, contiguous)
// ---------------------------------------------------------------------------
__global__ __launch_bounds__(256)
void build_layouts(const float* __restrict__ qkvv, const float* __restrict__ inv,
                   unsigned short* __restrict__ qn_t,  unsigned short* __restrict__ kn_t,
                   unsigned short* __restrict__ k_t,   unsigned short* __restrict__ vsa_t,
                   unsigned short* __restrict__ qn_tok, unsigned short* __restrict__ vca_tok)
{
    long long i = (long long)blockIdx.x * 256 + threadIdx.x;   // over B*H*N*D
    const int dd = (int)(i % DD); long long j = i / DD;
    const int n  = (int)(j % NN); j /= NN;
    const int h  = (int)(j % HH);
    const int b  = (int)(j / HH);

    const long long base = ((((long long)b * NN + n) * 4) * HH + h) * DD + dd;
    const long long cs = (long long)HH * DD;                   // per-component stride
    const float q   = qkvv[base];
    const float k   = qkvv[base + cs];
    const float vca = qkvv[base + 2 * cs];
    const float vsa = qkvv[base + 3 * cs];

    const float iq = inv[((0 * BB + b) * HH + h) * DD + dd];
    const float ik = inv[((1 * BB + b) * HH + h) * DD + dd];

    const int bh = b * HH + h;
    const long long ti = ((long long)bh * DD + dd) * NN + n;
    qn_t[ti]  = f32_bf16(q * iq);
    kn_t[ti]  = f32_bf16(k * ik);
    k_t[ti]   = f32_bf16(k);
    vsa_t[ti] = f32_bf16(vsa);

    const long long tok = ((long long)bh * NN + n) * DD + dd;
    qn_tok[tok]  = f32_bf16(q * iq);
    vca_tok[tok] = f32_bf16(vca);
}

// ---------------------------------------------------------------------------
// softmax over rows of length 64 (one wave32 per row, 2 elems/lane), with
// per-head temperature; emits bf16 probabilities for the following GEMM.
// ---------------------------------------------------------------------------
__global__ __launch_bounds__(256)
void softmax64_bf16(const float* __restrict__ S, unsigned short* __restrict__ out,
                    const float* __restrict__ temp, int rowsPerBatch, long long totalRows)
{
    const long long row = (long long)blockIdx.x * 8 + (threadIdx.x >> 5);
    const int lane = threadIdx.x & 31;
    if (row >= totalRows) return;

    const int bh = (int)(row / rowsPerBatch);
    const float tp = temp[bh % HH];

    const float* p = S + row * 64;
    float v0 = p[lane] * tp;
    float v1 = p[lane + 32] * tp;

    float m = fmaxf(v0, v1);
    #pragma unroll
    for (int o = 16; o > 0; o >>= 1) m = fmaxf(m, __shfl_xor(m, o, 32));
    const float e0 = __expf(v0 - m);
    const float e1 = __expf(v1 - m);
    float s = e0 + e1;
    #pragma unroll
    for (int o = 16; o > 0; o >>= 1) s += __shfl_xor(s, o, 32);
    const float r = 1.f / s;

    out[row * 64 + lane]      = f32_bf16(e0 * r);
    out[row * 64 + lane + 32] = f32_bf16(e1 * r);
}

// ---------------------------------------------------------------------------
// pack E-projection results: kproj[bh][d][p] -> kprojT bf16 [bh][p][d],
// vproj -> bf16 same-layout.
// ---------------------------------------------------------------------------
__global__ __launch_bounds__(256)
void proj_pack(const float* __restrict__ kproj, const float* __restrict__ vproj,
               unsigned short* __restrict__ kprojT, unsigned short* __restrict__ vprojB)
{
    const int i = blockIdx.x * 256 + threadIdx.x;     // over 32*64*64
    const int p  = i % PP;
    const int j  = i / PP;
    const int dd = j % DD;
    const int bh = j / DD;
    kprojT[((long long)bh * PP + p) * DD + dd] = f32_bf16(kproj[i]);
    vprojB[i] = f32_bf16(vproj[i]);
}

// ---------------------------------------------------------------------------
extern "C" void kernel_launch(void* const* d_in, const int* in_sizes, int n_in,
                              void* d_out, int out_size, void* d_ws, size_t ws_size,
                              hipStream_t stream)
{
    (void)in_sizes; (void)n_in; (void)out_size; (void)ws_size;

    const float* x      = (const float*)d_in[0];
    const float* Wqkvv  = (const float*)d_in[2];
    const float* W_E    = (const float*)d_in[3];
    const float* b_E    = (const float*)d_in[4];
    const float* temp1  = (const float*)d_in[5];
    const float* temp2  = (const float*)d_in[6];
    const float* W_out1 = (const float*)d_in[7];
    const float* b_out1 = (const float*)d_in[8];
    const float* W_out2 = (const float*)d_in[9];
    const float* b_out2 = (const float*)d_in[10];
    const float* prelu  = (const float*)d_in[11];
    float* out = (float*)d_out;

    char* w = (char*)d_ws;
    auto alloc = [&](size_t bytes) -> void* {
        void* p = (void*)w;
        w += (bytes + 255) & ~(size_t)255;
        return p;
    };

    const long long NE_TOK = (long long)MT * CC;          // 8.4M
    unsigned short* xb    = (unsigned short*)alloc(NE_TOK * 2);
    unsigned short* wqb   = (unsigned short*)alloc((size_t)4 * CC * CC * 2);
    unsigned short* web   = (unsigned short*)alloc((size_t)PP * NN * 2);
    unsigned short* wo1b  = (unsigned short*)alloc((size_t)(CC / 2) * CC * 2);
    unsigned short* wo2b  = (unsigned short*)alloc((size_t)(CC / 2) * CC * 2);
    float*          qkvv  = (float*)alloc((size_t)MT * 4 * CC * 4);
    float*          invn  = (float*)alloc((size_t)2 * BB * HH * DD * 4);
    const long long THD   = (long long)BB * HH * DD * NN;  // 8.4M
    unsigned short* qn_t  = (unsigned short*)alloc(THD * 2);
    unsigned short* kn_t  = (unsigned short*)alloc(THD * 2);
    unsigned short* k_t   = (unsigned short*)alloc(THD * 2);
    unsigned short* vsa_t = (unsigned short*)alloc(THD * 2);
    unsigned short* qn_tok  = (unsigned short*)alloc(THD * 2);
    unsigned short* vca_tok = (unsigned short*)alloc(THD * 2);
    float*          Sca   = (float*)alloc((size_t)32 * DD * DD * 4);
    unsigned short* attnC = (unsigned short*)alloc((size_t)32 * DD * DD * 2);
    float*          xca   = (float*)alloc(NE_TOK * 4);
    float*          kproj = (float*)alloc((size_t)32 * DD * PP * 4);
    float*          vproj = (float*)alloc((size_t)32 * DD * PP * 4);
    unsigned short* kprojT = (unsigned short*)alloc((size_t)32 * DD * PP * 2);
    unsigned short* vprojB = (unsigned short*)alloc((size_t)32 * DD * PP * 2);
    float*          scores = (float*)alloc((size_t)32 * NN * PP * 4);
    unsigned short* attnS  = (unsigned short*)alloc((size_t)32 * NN * PP * 2);
    float*          xsa   = (float*)alloc(NE_TOK * 4);
    unsigned short* xsab  = (unsigned short*)alloc(NE_TOK * 2);
    unsigned short* xcab  = (unsigned short*)alloc(NE_TOK * 2);

    dim3 blk(256);
    auto cg = [](long long n) { return dim3((unsigned)((n + 255) / 256)); };
    auto mg = [](int m) { return (unsigned)((m + 127) / 128); };   // GEMM grid.x

    // 0) quantize activations + weights to bf16
    cvt_f32_bf16<<<cg(NE_TOK), blk, 0, stream>>>(x, xb, NE_TOK);
    cvt_f32_bf16<<<cg(4 * CC * CC), blk, 0, stream>>>(Wqkvv, wqb, 4 * CC * CC);
    cvt_f32_bf16<<<cg((long long)PP * NN), blk, 0, stream>>>(W_E, web, (long long)PP * NN);
    cvt_f32_bf16<<<cg((CC / 2) * CC), blk, 0, stream>>>(W_out1, wo1b, (CC / 2) * CC);
    cvt_f32_bf16<<<cg((CC / 2) * CC), blk, 0, stream>>>(W_out2, wo2b, (CC / 2) * CC);

    // 1) qkvv = x @ W_qkvv^T            [32768 x 1024], K=256
    gemm_bf16_wmma<<<dim3(mg(MT), (4 * CC) / 64, 1), blk, 0, stream>>>(
        xb, 0, CC, wqb, 0, CC,
        qkvv, 0, 0, 1, 4 * CC, 0, nullptr, nullptr, MT, 4 * CC, CC);

    // 2) per-channel inverse L2 norms for q,k
    qk_invnorm<<<dim3(2 * BB * HH * DD), blk, 0, stream>>>(qkvv, invn);

    // 3) build bf16 operand layouts
    build_layouts<<<cg(THD), blk, 0, stream>>>(qkvv, invn, qn_t, kn_t, k_t, vsa_t,
                                               qn_tok, vca_tok);

    // 4) channel-attn scores: S[bh] = qn_t[bh] (64xN) . kn_t[bh]^T   K=4096
    gemm_bf16_wmma<<<dim3(mg(DD), 1, 32), blk, 0, stream>>>(
        qn_t, (long long)DD * NN, NN, kn_t, (long long)DD * NN, NN,
        Sca, (long long)DD * DD, 0, 1, DD, 0, nullptr, nullptr, DD, DD, NN);

    // 5) softmax rows (temperature) -> bf16 attn_ca
    softmax64_bf16<<<dim3((32 * DD) / 8), blk, 0, stream>>>(Sca, attnC, temp1, DD,
                                                            (long long)32 * DD);

    // 6) x_ca[bh][n][d] = vca_tok . attn_ca^T -> scattered into [B,N,C]
    gemm_bf16_wmma<<<dim3(mg(NN), 1, 32), blk, 0, stream>>>(
        vca_tok, (long long)NN * DD, DD, attnC, (long long)DD * DD, DD,
        xca, (long long)NN * CC, DD, HH, CC, 0, nullptr, nullptr, NN, DD, DD);

    // 7) E-projections: k_proj / v_sa_proj = {k,vsa}_t (64xN) . W_E^T  K=4096
    gemm_bf16_wmma<<<dim3(mg(DD), 1, 32), blk, 0, stream>>>(
        k_t, (long long)DD * NN, NN, web, 0, NN,
        kproj, (long long)DD * PP, 0, 1, PP, 0, b_E, nullptr, DD, PP, NN);
    gemm_bf16_wmma<<<dim3(mg(DD), 1, 32), blk, 0, stream>>>(
        vsa_t, (long long)DD * NN, NN, web, 0, NN,
        vproj, (long long)DD * PP, 0, 1, PP, 0, b_E, nullptr, DD, PP, NN);

    // 8) pack projections to bf16 (kproj transposed to [P][d])
    proj_pack<<<cg((long long)32 * DD * PP), blk, 0, stream>>>(kproj, vproj, kprojT, vprojB);

    // 9) spatial scores: [bh][n][p] = qn_tok . kprojT^T   K=64
    gemm_bf16_wmma<<<dim3(mg(NN), 1, 32), blk, 0, stream>>>(
        qn_tok, (long long)NN * DD, DD, kprojT, (long long)PP * DD, DD,
        scores, (long long)NN * PP, 0, 1, PP, 0, nullptr, nullptr, NN, PP, DD);

    // 10) softmax over P (temperature2) -> bf16 attn_sa
    softmax64_bf16<<<dim3((32 * NN) / 8), blk, 0, stream>>>(scores, attnS, temp2, NN,
                                                            (long long)32 * NN);

    // 11) x_sa[bh][n][d] = attn_sa . vproj^T -> scattered into [B,N,C]
    gemm_bf16_wmma<<<dim3(mg(NN), 1, 32), blk, 0, stream>>>(
        attnS, (long long)NN * PP, PP, vprojB, (long long)DD * PP, PP,
        xsa, (long long)NN * CC, DD, HH, CC, 0, nullptr, nullptr, NN, DD, PP);

    // 12) bf16 copies of branch activations
    cvt_f32_bf16<<<cg(NE_TOK), blk, 0, stream>>>(xsa, xsab, NE_TOK);
    cvt_f32_bf16<<<cg(NE_TOK), blk, 0, stream>>>(xca, xcab, NE_TOK);

    // 13) out[:, :128] = PReLU(x_sa @ W_out1^T + b1); out[:, 128:] = PReLU(x_ca @ W_out2^T + b2)
    gemm_bf16_wmma<<<dim3(mg(MT), (CC / 2) / 64, 1), blk, 0, stream>>>(
        xsab, 0, CC, wo1b, 0, CC,
        out, 0, 0, 1, CC, 0, b_out1, prelu, MT, CC / 2, CC);
    gemm_bf16_wmma<<<dim3(mg(MT), (CC / 2) / 64, 1), blk, 0, stream>>>(
        xcab, 0, CC, wo2b, 0, CC,
        out, 0, 0, 1, CC, CC / 2, b_out2, prelu, MT, CC / 2, CC);
}
